// NewPool_20839181320911
// MI455X (gfx1250) — matrix operand order repaired and checked by reading, not compiled, not run on hardware
//
#include <hip/hip_runtime.h>

#define F 96
#define EPSF 1e-15f
#define PROJ_EPS 4e-3f

typedef __attribute__((ext_vector_type(16))) _Float16 v16h;
typedef __attribute__((ext_vector_type(8)))  _Float16 v8h;
typedef __attribute__((ext_vector_type(8)))  float    v8f;

__device__ __forceinline__ float wave_sum(float v) {
#pragma unroll
    for (int off = 16; off > 0; off >>= 1) v += __shfl_xor(v, off, 32);
    return v;
}

// ---------------- logmap0: x_tan = x/||x|| * atanh(||x||), also emit f16 copy -------------
__global__ void logmap_kernel(const float* __restrict__ x, float* __restrict__ xt,
                              _Float16* __restrict__ xth, int N) {
    int wave = (blockIdx.x * blockDim.x + threadIdx.x) >> 5;
    int lane = threadIdx.x & 31;
    if (wave >= N) return;
    const float* xr = x + (size_t)wave * F;
    float f0 = xr[lane], f1 = xr[lane + 32], f2 = xr[lane + 64];
    float ss = wave_sum(f0 * f0 + f1 * f1 + f2 * f2);
    float nrm = fmaxf(sqrtf(ss), EPSF);
    float arg = fminf(nrm, 1.0f - EPSF);
    float s = atanhf(arg) / nrm;
    float g0 = f0 * s, g1 = f1 * s, g2 = f2 * s;
    float* o = xt + (size_t)wave * F;
    _Float16* oh = xth + (size_t)wave * F;
    o[lane] = g0; o[lane + 32] = g1; o[lane + 64] = g2;
    oh[lane] = (_Float16)g0; oh[lane + 32] = (_Float16)g1; oh[lane + 64] = (_Float16)g2;
}

// ---------------- degree of source nodes ----------------
__global__ void deg_kernel(const int* __restrict__ row, float* __restrict__ deg, int E) {
    int e = blockIdx.x * blockDim.x + threadIdx.x;
    if (e < E) atomicAdd(&deg[row[e]], 1.0f);
}

__global__ void dinv_kernel(const float* __restrict__ deg, float* __restrict__ dinv, int N) {
    int i = blockIdx.x * blockDim.x + threadIdx.x;
    if (i < N) { float d = deg[i]; dinv[i] = (d > 0.0f) ? rsqrtf(d) : 0.0f; }
}

// ---------------- info scatter: info[col] += -dinv[row]*dinv[col]*x_tan[row] ----------------
__global__ void info_scatter(const int* __restrict__ row, const int* __restrict__ col,
                             const float* __restrict__ dinv, const float* __restrict__ xt,
                             float* __restrict__ info, int E) {
    int e = (blockIdx.x * blockDim.x + threadIdx.x) >> 5;
    int lane = threadIdx.x & 31;
    if (e >= E) return;
    int r = row[e], c = col[e];
    float coef = -dinv[r] * dinv[c];
    const float* src = xt + (size_t)r * F;
    float* dst = info + (size_t)c * F;
    atomicAdd(&dst[lane],      coef * src[lane]);
    atomicAdd(&dst[lane + 32], coef * src[lane + 32]);
    atomicAdd(&dst[lane + 64], coef * src[lane + 64]);
}

// ---------------- score = sum |info| ; sel = score > 1 ----------------
__global__ void score_kernel(const float* __restrict__ info, float* __restrict__ sel, int N) {
    int wave = (blockIdx.x * blockDim.x + threadIdx.x) >> 5;
    int lane = threadIdx.x & 31;
    if (wave >= N) return;
    const float* p = info + (size_t)wave * F;
    float s = wave_sum(fabsf(p[lane]) + fabsf(p[lane + 32]) + fabsf(p[lane + 64]));
    if (lane == 0) sel[wave] = (s > 1.0f) ? 1.0f : 0.0f;
}

// ---------------- W_up f32 -> f16 ----------------
__global__ void convw_kernel(const float* __restrict__ w, _Float16* __restrict__ wh, int n) {
    int i = blockIdx.x * blockDim.x + threadIdx.x;
    if (i < n) wh[i] = (_Float16)w[i];
}

// ---------------- WMMA GEMM: updated = leaky_relu(x_tan @ W_up^T) ----------------
// A: x_tan_h [Npad, 96] row-major (M x K), pad rows zeroed.
// B[k,n] = W_up[n,k]; W_up_h row-major [96,96].
// One wave computes one 16x16 D tile; 3 WMMAs over K=96 per n-tile; 6 n-tiles.
// Output buffer is padded to Npad rows -> unconditional stores (no exec-mask churn).
__global__ void gemm_wmma(const _Float16* __restrict__ A, const _Float16* __restrict__ Wh,
                          float* __restrict__ out, int rowTiles) {
    int tile = blockIdx.x * (blockDim.x >> 5) + (threadIdx.x >> 5);
    if (tile >= rowTiles) return;
    int lane = threadIdx.x & 31;
    int half = lane >> 4;     // 0: K 0..7 / 16..23 ; 1: K 8..15 / 24..31
    int mrow = lane & 15;

    // A fragments for the 3 K-chunks (16-bit A 16x32 layout)
    const _Float16* arow = A + ((size_t)tile * 16 + mrow) * F;
    v16h a[3];
#pragma unroll
    for (int kb = 0; kb < 3; ++kb) {
        v8h lo = *(const v8h*)(arow + kb * 32 + half * 8);
        v8h hi = *(const v8h*)(arow + kb * 32 + 16 + half * 8);
#pragma unroll
        for (int j = 0; j < 8; ++j) { a[kb][j] = lo[j]; a[kb][8 + j] = hi[j]; }
    }

    int ncol = lane & 15;
    int koff = half * 8;      // B 32x16 layout: lanes 0-15 hold K 0..7 / 16..23, lanes 16-31 hold K 8..15 / 24..31
    float* obase = out + (size_t)tile * 16 * F + ((size_t)half * 8) * F + ncol;
#pragma unroll
    for (int nt = 0; nt < 6; ++nt) {
        const _Float16* wrow = Wh + (size_t)(nt * 16 + ncol) * F; // B[:,n] = W_up row n
        v8f c = {};
#pragma unroll
        for (int kb = 0; kb < 3; ++kb) {
            v8h b0 = *(const v8h*)(wrow + kb * 32 + koff);
            v8h b1 = *(const v8h*)(wrow + kb * 32 + koff + 16);
            v16h b;
#pragma unroll
            for (int j = 0; j < 8; ++j) { b[j] = b0[j]; b[8 + j] = b1[j]; }
            c = __builtin_amdgcn_wmma_f32_16x16x32_f16(false, a[kb], false, b,
                                                       (short)0, c, false, false);
        }
        // D layout: VGPR v -> M = v + 8*half, N = lane&15. Padded buffer: no guards.
#pragma unroll
        for (int v = 0; v < 8; ++v) {
            float val = c[v];
            obase[(size_t)v * F + nt * 16] = (val > 0.0f) ? val : 0.01f * val;
        }
    }
}

// ---------------- neighbor sums: sum_neigh[col] += updated[row]; sum_sel likewise if sel ----
__global__ void neigh_kernel(const int* __restrict__ row, const int* __restrict__ col,
                             const float* __restrict__ sel, const float* __restrict__ upd,
                             float* __restrict__ sn, float* __restrict__ ssl, int E) {
    int e = (blockIdx.x * blockDim.x + threadIdx.x) >> 5;
    int lane = threadIdx.x & 31;
    if (e >= E) return;
    int r = row[e], c = col[e];
    const float* src = upd + (size_t)r * F;
    __builtin_prefetch(src, 0, 0);
    float s = sel[r];
    float u0 = src[lane], u1 = src[lane + 32], u2 = src[lane + 64];
    float* d0 = sn + (size_t)c * F;
    atomicAdd(&d0[lane], u0); atomicAdd(&d0[lane + 32], u1); atomicAdd(&d0[lane + 64], u2);
    if (s > 0.0f) {
        float* d1 = ssl + (size_t)c * F;
        atomicAdd(&d1[lane], u0); atomicAdd(&d1[lane + 32], u1); atomicAdd(&d1[lane + 64], u2);
    }
}

// ---------------- w_sel = sigmoid(concat(sum_sel,sum_neigh) @ W_lw^T); selw = w_sel*sel ----
__global__ void wsel_kernel(const float* __restrict__ ssl, const float* __restrict__ sn,
                            const float* __restrict__ Wlw, const float* __restrict__ sel,
                            float* __restrict__ selw, int N) {
    int wave = (blockIdx.x * blockDim.x + threadIdx.x) >> 5;
    int lane = threadIdx.x & 31;
    if (wave >= N) return;
    const float* a = ssl + (size_t)wave * F;
    const float* b = sn + (size_t)wave * F;
    float acc = 0.0f;
#pragma unroll
    for (int j = 0; j < 3; ++j) {
        int k = lane + 32 * j;
        acc += a[k] * Wlw[k] + b[k] * Wlw[F + k];
    }
    acc = wave_sum(acc);
    if (lane == 0) {
        float sig = 1.0f / (1.0f + expf(-acc));
        selw[wave] = sig * sel[wave];
    }
}

// ---------------- A_x scatter: Ax[col] += selw[row] * updated[row] ----------------
__global__ void ax_kernel(const int* __restrict__ row, const int* __restrict__ col,
                          const float* __restrict__ selw, const float* __restrict__ upd,
                          float* __restrict__ Ax, int E) {
    int e = (blockIdx.x * blockDim.x + threadIdx.x) >> 5;
    int lane = threadIdx.x & 31;
    if (e >= E) return;
    int r = row[e];
    float cw = selw[r];
    if (cw == 0.0f) return;
    int c = col[e];
    const float* src = upd + (size_t)r * F;
    float* dst = Ax + (size_t)c * F;
    atomicAdd(&dst[lane],      cw * src[lane]);
    atomicAdd(&dst[lane + 32], cw * src[lane + 32]);
    atomicAdd(&dst[lane + 64], cw * src[lane + 64]);
}

// ---------------- out = proj(expmap0(updated + relu(Ax))) ----------------
__global__ void finalize_kernel(const float* __restrict__ upd, const float* __restrict__ Ax,
                                float* __restrict__ out, int N) {
    int wave = (blockIdx.x * blockDim.x + threadIdx.x) >> 5;
    int lane = threadIdx.x & 31;
    if (wave >= N) return;
    const float* u = upd + (size_t)wave * F;
    const float* a = Ax + (size_t)wave * F;
    float o0 = u[lane]      + fmaxf(a[lane], 0.0f);
    float o1 = u[lane + 32] + fmaxf(a[lane + 32], 0.0f);
    float o2 = u[lane + 64] + fmaxf(a[lane + 64], 0.0f);
    float ss = wave_sum(o0 * o0 + o1 * o1 + o2 * o2);
    float nrm = fmaxf(sqrtf(ss), EPSF);
    float s = tanhf(nrm) / nrm;                 // expmap0 scale (c=1)
    float ynorm = fmaxf(nrm * s, EPSF);         // ||expmap0(o)||
    float maxnorm = 1.0f - PROJ_EPS;
    if (ynorm > maxnorm) s *= maxnorm / ynorm;  // Poincare-ball projection
    float* o = out + (size_t)wave * F;
    o[lane] = o0 * s; o[lane + 32] = o1 * s; o[lane + 64] = o2 * s;
}

extern "C" void kernel_launch(void* const* d_in, const int* in_sizes, int n_in,
                              void* d_out, int out_size, void* d_ws, size_t ws_size,
                              hipStream_t stream) {
    (void)n_in; (void)out_size; (void)ws_size;
    const float* x    = (const float*)d_in[0];
    const int*   ei   = (const int*)d_in[1];
    const float* W_up = (const float*)d_in[2];
    const float* W_lw = (const float*)d_in[3];
    const int N = in_sizes[0] / F;
    const int E = in_sizes[1] / 2;
    const int* row = ei;
    const int* col = ei + E;

    const int rowTiles = (N + 15) / 16;
    const int Npad = rowTiles * 16;

    char* p = (char*)d_ws;
    auto alloc = [&](size_t bytes) -> char* {
        char* r = p; p += (bytes + 255) & ~(size_t)255; return r;
    };
    float*    x_tan = (float*)alloc((size_t)N * F * 4);
    _Float16* xth   = (_Float16*)alloc((size_t)Npad * F * 2);
    float*    upd   = (float*)alloc((size_t)Npad * F * 4);   // padded: unguarded WMMA stores
    float*    info  = (float*)alloc((size_t)N * F * 4);
    float*    sn    = (float*)alloc((size_t)N * F * 4);
    float*    ssl   = (float*)alloc((size_t)N * F * 4);
    float*    Ax    = (float*)alloc((size_t)N * F * 4);
    float*    deg   = (float*)alloc((size_t)N * 4);
    float*    dinv  = (float*)alloc((size_t)N * 4);
    float*    sel   = (float*)alloc((size_t)N * 4);
    float*    selw  = (float*)alloc((size_t)N * 4);
    _Float16* Wh    = (_Float16*)alloc((size_t)F * F * 2);

    // zero accumulators every call (graph-capture-safe)
    hipMemsetAsync(xth, 0, (size_t)Npad * F * 2, stream);
    hipMemsetAsync(deg, 0, (size_t)N * 4, stream);
    hipMemsetAsync(sn,  0, (size_t)N * F * 4, stream);
    hipMemsetAsync(ssl, 0, (size_t)N * F * 4, stream);
    hipMemsetAsync(Ax,  0, (size_t)N * F * 4, stream);

    const int blk = 256;                       // 8 wave32s per block
    const int nodeBlocks = (N + 7) / 8;
    const int edgeBlocks = (E + 7) / 8;

    logmap_kernel<<<nodeBlocks, blk, 0, stream>>>(x, x_tan, xth, N);
    hipMemcpyAsync(info, x_tan, (size_t)N * F * 4, hipMemcpyDeviceToDevice, stream);
    deg_kernel<<<(E + 255) / 256, 256, 0, stream>>>(row, deg, E);
    dinv_kernel<<<(N + 255) / 256, 256, 0, stream>>>(deg, dinv, N);
    info_scatter<<<edgeBlocks, blk, 0, stream>>>(row, col, dinv, x_tan, info, E);
    score_kernel<<<nodeBlocks, blk, 0, stream>>>(info, sel, N);
    convw_kernel<<<(F * F + 255) / 256, 256, 0, stream>>>(W_up, Wh, F * F);
    gemm_wmma<<<(rowTiles + 7) / 8, blk, 0, stream>>>(xth, Wh, upd, rowTiles);
    neigh_kernel<<<edgeBlocks, blk, 0, stream>>>(row, col, sel, upd, sn, ssl, E);
    wsel_kernel<<<nodeBlocks, blk, 0, stream>>>(ssl, sn, W_lw, sel, selw, N);
    ax_kernel<<<edgeBlocks, blk, 0, stream>>>(row, col, selw, upd, Ax, E);
    finalize_kernel<<<nodeBlocks, blk, 0, stream>>>(upd, Ax, (float*)d_out, N);
}
